// SimpleNN_53480932769918
// MI455X (gfx1250) — matrix operand order repaired and checked by reading, not compiled, not run on hardware
//
#include <hip/hip_runtime.h>
#include <math.h>

// ---------------------------------------------------------------------------
// CDNA5 (gfx1250) WMMA types
// ---------------------------------------------------------------------------
typedef __attribute__((ext_vector_type(16))) __bf16 v16bf;
typedef __attribute__((ext_vector_type(8)))  float  v8f;

union BF16Frag {
    v16bf v;
    uint4 q[2];
};

union PackOut {
    __bf16 b[16];
    uint4  q[2];
};

static __device__ __forceinline__ v8f wmma_bf16(v16bf a, v16bf b, v8f c) {
    // (neg_a, A, neg_b, B, c_mod, C, reuse_a, reuse_b)
    return __builtin_amdgcn_wmma_f32_16x16x32_bf16(false, a, false, b, (short)0, c, false, false);
}

// ---------------------------------------------------------------------------
// 1) Lorenz RK4 table: dyn(m) for m = i - 1024, i in [0, 2048)
//    Matches the reference fp32 RK4 (dt = 1/100, 100 steps).
// ---------------------------------------------------------------------------
static __device__ __forceinline__ void lorenz_deriv(float x, float y, float z,
                                                    float& dx, float& dy, float& dz) {
    dx = 10.0f * (y - x);
    dy = x * (28.0f - z) - y;
    dz = x * y - (8.0f / 3.0f) * z;
}

__global__ void lorenz_table_kern(float* __restrict__ table) {
    int i = blockIdx.x * blockDim.x + threadIdx.x;
    if (i >= 2048) return;
    float m = (float)(i - 1024);
    float x = m, y = m + 1.0f, z = 0.0f;
    const float dt  = 0.01f;
    const float hdt = 0.5f * dt;
    const float dt6 = dt / 6.0f;
    for (int s = 0; s < 100; ++s) {
        float k1x, k1y, k1z, k2x, k2y, k2z, k3x, k3y, k3z, k4x, k4y, k4z;
        lorenz_deriv(x, y, z, k1x, k1y, k1z);
        float ax = x + hdt * k1x, ay = y + hdt * k1y, az = z + hdt * k1z;
        lorenz_deriv(ax, ay, az, k2x, k2y, k2z);
        float bx = x + hdt * k2x, by = y + hdt * k2y, bz = z + hdt * k2z;
        lorenz_deriv(bx, by, bz, k3x, k3y, k3z);
        float cx = x + dt * k3x, cy = y + dt * k3y, cz = z + dt * k3z;
        lorenz_deriv(cx, cy, cz, k4x, k4y, k4z);
        x = x + dt6 * (k1x + 2.0f * k2x + 2.0f * k3x + k4x);
        y = y + dt6 * (k1y + 2.0f * k2y + 2.0f * k3y + k4y);
        z = z + dt6 * (k1z + 2.0f * k2z + 2.0f * k3z + k4z);
    }
    table[i] = x;
}

// ---------------------------------------------------------------------------
// Packed-fragment layout:
//   A operand [M x K]: tiles of 16(M) x 32(K). tile id = mt*(K/32) + kt.
//     Each tile = 1 KB: lane l owns 16 bf16 at byte offset l*32.
//       lanes 0-15 : row = l,     elements 0-7 = K[kt*32 + 0..8),  8-15 = K[+16..24)
//       lanes 16-31: row = l-16,  elements 0-7 = K[kt*32 + 8..16), 8-15 = K[+24..32)
//   B operand [K x N]: tiles of 32(K) x 16(N). tile id = nt*(K/32) + kt.
//     lane l = N col (nt*16 + l%16); elements 0-15 = K rows in the same order as A.
// This is exactly the VGPR layout v_wmma_f32_16x16x32_bf16 expects, so the
// GEMM streams contiguous 1 KB tiles (two b128 per lane at tile + lane*32).
// ---------------------------------------------------------------------------

// 2a) Pack A operand: f32 [M x K] -> bf16 hi/lo packed fragments.
__global__ void pack_a_kern(const float* __restrict__ src,
                            __bf16* __restrict__ hi, __bf16* __restrict__ lo,
                            int K, int ntile) {
    int tid  = blockIdx.x * blockDim.x + threadIdx.x;
    int lane = tid & 31;
    int tile = tid >> 5;
    if (tile >= ntile) return;
    const int KT = K >> 5;
    const int mt = tile / KT;
    const int kt = tile - mt * KT;
    const int row   = mt * 16 + (lane & 15);
    const int kbase = kt * 32 + ((lane >> 4) << 3);

    const float* p0 = src + (size_t)row * K + kbase;       // 8 floats
    const float* p1 = p0 + 16;                             // 8 floats
    float f[16];
    *(float4*)(f + 0)  = *(const float4*)(p0);
    *(float4*)(f + 4)  = *(const float4*)(p0 + 4);
    *(float4*)(f + 8)  = *(const float4*)(p1);
    *(float4*)(f + 12) = *(const float4*)(p1 + 4);

    PackOut H, L;
#pragma unroll
    for (int r = 0; r < 16; ++r) {
        __bf16 h = (__bf16)f[r];
        H.b[r] = h;
        L.b[r] = (__bf16)(f[r] - (float)h);
    }
    size_t o = (size_t)tile * 512 + lane * 16;
    *(uint4*)(hi + o) = H.q[0];
    *(uint4*)(hi + o + 8) = H.q[1];
    *(uint4*)(lo + o) = L.q[0];
    *(uint4*)(lo + o + 8) = L.q[1];
}

// 2b) Pack B operand: f32 [K x N] -> bf16 hi/lo packed fragments.
__global__ void pack_b_kern(const float* __restrict__ src,
                            __bf16* __restrict__ hi, __bf16* __restrict__ lo,
                            int K, int N, int ntile) {
    int tid  = blockIdx.x * blockDim.x + threadIdx.x;
    int lane = tid & 31;
    int tile = tid >> 5;
    if (tile >= ntile) return;
    const int KT = K >> 5;
    const int nt = tile / KT;
    const int kt = tile - nt * KT;
    const int row = kt * 32 + ((lane >> 4) << 3);
    const int col = nt * 16 + (lane & 15);

    const float* p = src + (size_t)row * N + col;          // K rows row+0..8 and row+16..24, column col
    float f[16];
#pragma unroll
    for (int r = 0; r < 8; ++r) {
        f[r]     = p[(size_t)r * N];
        f[r + 8] = p[(size_t)(r + 16) * N];
    }

    PackOut H, L;
#pragma unroll
    for (int r = 0; r < 16; ++r) {
        __bf16 h = (__bf16)f[r];
        H.b[r] = h;
        L.b[r] = (__bf16)(f[r] - (float)h);
    }
    size_t o = (size_t)tile * 512 + lane * 16;
    *(uint4*)(hi + o) = H.q[0];
    *(uint4*)(hi + o + 8) = H.q[1];
    *(uint4*)(lo + o) = L.q[0];
    *(uint4*)(lo + o + 8) = L.q[1];
}

// 3) knot_gelu (table lookup) on h [M x K], emitting packed A fragments for GEMM2.
__global__ void act_pack_kern(const float* __restrict__ hsrc, const float* __restrict__ table,
                              __bf16* __restrict__ hi, __bf16* __restrict__ lo,
                              int K, int ntile) {
    int tid  = blockIdx.x * blockDim.x + threadIdx.x;
    int lane = tid & 31;
    int tile = tid >> 5;
    if (tile >= ntile) return;
    const int KT = K >> 5;
    const int mt = tile / KT;
    const int kt = tile - mt * KT;
    const int row   = mt * 16 + (lane & 15);
    const int kbase = kt * 32 + ((lane >> 4) << 3);

    const float* p0 = hsrc + (size_t)row * K + kbase;
    const float* p1 = p0 + 16;
    float f[16];
    *(float4*)(f + 0)  = *(const float4*)(p0);
    *(float4*)(f + 4)  = *(const float4*)(p0 + 4);
    *(float4*)(f + 8)  = *(const float4*)(p1);
    *(float4*)(f + 12) = *(const float4*)(p1 + 4);

    const float c = 0.7978845608028654f; // sqrt(2/pi)
    PackOut H, L;
#pragma unroll
    for (int r = 0; r < 16; ++r) {
        float x = f[r];
        int ti = (int)ceilf(x) + 1024;
        ti = ti < 0 ? 0 : (ti > 2047 ? 2047 : ti);
        float dyn = table[ti];
        float inner = c * (x + 0.044715f * x * x * x);
        float a = 0.5f * x * (1.0f + tanhf(inner) * dyn);
        __bf16 h = (__bf16)a;
        H.b[r] = h;
        L.b[r] = (__bf16)(a - (float)h);
    }
    size_t o = (size_t)tile * 512 + lane * 16;
    *(uint4*)(hi + o) = H.q[0];
    *(uint4*)(hi + o + 8) = H.q[1];
    *(uint4*)(lo + o) = L.q[0];
    *(uint4*)(lo + o + 8) = L.q[1];
}

// ---------------------------------------------------------------------------
// 4) bf16x3 split GEMM on packed fragments.
//    C[M,N] = (Ahi+Alo) @ (Bhi+Blo) + bias, f32 accumulate, f32 output.
//    Block = 256 threads = 8 waves (2 M-waves x 4 N-waves).
//    Block tile 64(M) x 128(N); wave tile 32x32 = 2x2 WMMA frags.
//    Requires: M%64==0, N%128==0, K%32==0.
// ---------------------------------------------------------------------------
__global__ __launch_bounds__(256) void gemm_bf16x3_kern(
    const __bf16* __restrict__ Ahi, const __bf16* __restrict__ Alo,
    const __bf16* __restrict__ Bhi, const __bf16* __restrict__ Blo,
    const float* __restrict__ bias, float* __restrict__ C,
    int M, int N, int K)
{
    const int lane = threadIdx.x & 31;
    const int wave = threadIdx.x >> 5;
    const int wm = wave >> 2;               // 0..1
    const int wn = wave & 3;                // 0..3
    const int m_wave = blockIdx.y * 64 + wm * 32;
    const int n_wave = blockIdx.x * 128 + wn * 32;

    const int KT = K >> 5;
    const size_t laneoff = (size_t)lane * 16;

    // Per-(frag-row/col) tile streams; inner K loop walks contiguous 1KB tiles.
    const __bf16* pAh[2], *pAl[2], *pBh[2], *pBl[2];
#pragma unroll
    for (int i = 0; i < 2; ++i) {
        size_t base = (size_t)(m_wave / 16 + i) * KT * 512 + laneoff;
        pAh[i] = Ahi + base;
        pAl[i] = Alo + base;
    }
#pragma unroll
    for (int j = 0; j < 2; ++j) {
        size_t base = (size_t)(n_wave / 16 + j) * KT * 512 + laneoff;
        pBh[j] = Bhi + base;
        pBl[j] = Blo + base;
    }

    v8f acc[2][2] = {};

    for (int kt = 0; kt < KT; ++kt) {
        const size_t toff = (size_t)kt * 512;
        BF16Frag ah[2], al[2], bh[2], bl[2];
#pragma unroll
        for (int i = 0; i < 2; ++i) {
            ah[i].q[0] = *(const uint4*)(pAh[i] + toff);
            ah[i].q[1] = *(const uint4*)(pAh[i] + toff + 8);
            al[i].q[0] = *(const uint4*)(pAl[i] + toff);
            al[i].q[1] = *(const uint4*)(pAl[i] + toff + 8);
        }
#pragma unroll
        for (int j = 0; j < 2; ++j) {
            bh[j].q[0] = *(const uint4*)(pBh[j] + toff);
            bh[j].q[1] = *(const uint4*)(pBh[j] + toff + 8);
            bl[j].q[0] = *(const uint4*)(pBl[j] + toff);
            bl[j].q[1] = *(const uint4*)(pBl[j] + toff + 8);
        }
#pragma unroll
        for (int i = 0; i < 2; ++i)
#pragma unroll
            for (int j = 0; j < 2; ++j) {
                acc[i][j] = wmma_bf16(ah[i].v, bh[j].v, acc[i][j]);
                acc[i][j] = wmma_bf16(ah[i].v, bl[j].v, acc[i][j]);
                acc[i][j] = wmma_bf16(al[i].v, bh[j].v, acc[i][j]);
            }
    }

    // C layout: VGPR r -> M = r + (lane>=16 ? 8 : 0), N = lane & 15.
    const int ncol = lane & 15;
    const int moff = (lane >> 4) << 3;
#pragma unroll
    for (int j = 0; j < 2; ++j) {
        const int n = n_wave + 16 * j + ncol;
        const float bv = bias[n];
#pragma unroll
        for (int i = 0; i < 2; ++i) {
            const int mbase = m_wave + 16 * i + moff;
#pragma unroll
            for (int r = 0; r < 8; ++r) {
                C[(size_t)(mbase + r) * N + n] = acc[i][j][r] + bv;
            }
        }
    }
}

// ---------------------------------------------------------------------------
// Launch
// ---------------------------------------------------------------------------
extern "C" void kernel_launch(void* const* d_in, const int* in_sizes, int n_in,
                              void* d_out, int out_size, void* d_ws, size_t ws_size,
                              hipStream_t stream) {
    const float* x  = (const float*)d_in[0];
    const float* W1 = (const float*)d_in[1];
    const float* b1 = (const float*)d_in[2];
    const float* W2 = (const float*)d_in[3];
    const float* b2 = (const float*)d_in[4];
    float* out = (float*)d_out;

    const int Dh   = in_sizes[2];
    const int Dout = in_sizes[4];
    const int Din  = in_sizes[1] / Dh;
    const int Bm   = in_sizes[0] / Din;

    char* ws = (char*)d_ws;
    size_t off = 0;
    auto carve = [&](size_t bytes) -> char* {
        char* p = ws + off;
        off += (bytes + 255) & ~(size_t)255;
        return p;
    };
    float*  table = (float*)carve(2048 * sizeof(float));
    __bf16* xhi   = (__bf16*)carve((size_t)Bm * Din * 2);
    __bf16* xlo   = (__bf16*)carve((size_t)Bm * Din * 2);
    __bf16* w1hi  = (__bf16*)carve((size_t)Din * Dh * 2);
    __bf16* w1lo  = (__bf16*)carve((size_t)Din * Dh * 2);
    __bf16* w2hi  = (__bf16*)carve((size_t)Dh * Dout * 2);
    __bf16* w2lo  = (__bf16*)carve((size_t)Dh * Dout * 2);
    float*  hbuf  = (float*)carve((size_t)Bm * Dh * 4);
    __bf16* ahi   = (__bf16*)carve((size_t)Bm * Dh * 2);
    __bf16* alo   = (__bf16*)carve((size_t)Bm * Dh * 2);

    // 1) Lorenz dyn(m) lookup table
    lorenz_table_kern<<<8, 256, 0, stream>>>(table);

    // 2) Pack operands into WMMA fragment layout (bf16 hi/lo)
    {
        int nt = (Bm / 16) * (Din / 32);                 // x as A of GEMM1
        pack_a_kern<<<(nt * 32 + 255) / 256, 256, 0, stream>>>(x, xhi, xlo, Din, nt);
    }
    {
        int nt = (Dh / 16) * (Din / 32);                 // W1 as B of GEMM1
        pack_b_kern<<<(nt * 32 + 255) / 256, 256, 0, stream>>>(W1, w1hi, w1lo, Din, Dh, nt);
    }
    {
        int nt = (Dout / 16) * (Dh / 32);                // W2 as B of GEMM2
        pack_b_kern<<<(nt * 32 + 255) / 256, 256, 0, stream>>>(W2, w2hi, w2lo, Dh, Dout, nt);
    }

    // 3) GEMM1: h = x @ W1 + b1
    dim3 g1(Dh / 128, Bm / 64);
    gemm_bf16x3_kern<<<g1, 256, 0, stream>>>(xhi, xlo, w1hi, w1lo, b1, hbuf, Bm, Dh, Din);

    // 4) activation; emit GEMM2's A operand in packed fragment layout
    {
        int nt = (Bm / 16) * (Dh / 32);
        act_pack_kern<<<(nt * 32 + 255) / 256, 256, 0, stream>>>(hbuf, table, ahi, alo, Dh, nt);
    }

    // 5) GEMM2: out = a @ W2 + b2
    dim3 g2(Dout / 128, Bm / 64);
    gemm_bf16x3_kern<<<g2, 256, 0, stream>>>(ahi, alo, w2hi, w2lo, b2, out, Bm, Dout, Dh);
}
